// SSL_HSIC_84550726189308
// MI455X (gfx1250) — matrix operand order repaired and checked by reading, not compiled, not run on hardware
//
#include <hip/hip_runtime.h>
#include <hip/hip_bf16.h>
#include <math.h>

// ---------------------------------------------------------------------------
// HSIC objective for SSL (B=4096, M=2 views, N=8192, D=128, sigma=1, gamma=3)
//
// K[i,j] = exp(-(n_i + n_j - 2*x_i.x_j)/2) is never materialized (256 MB);
// all needed statistics are streaming reductions over 16x16 WMMA tiles:
//   pos   = sum K[i,j] over |i-j| == B   (only tiles with |i0-j0| == B)
//   sumK  = sum K              (off-diag total = sumK - N, diag K == 1)
//   sumK2 = sum K^2
//   s_i   = row sums           (mean(Kc^2) = sumK2/N^2 - 2*sum s^2/N^3 + t^2)
// Gram matrix in f16 WMMA with f32 accumulation (inputs are L2-normalized).
// Double-buffered B tiles: 2 tiles of loads in flight over each WMMA group.
// Raw v_exp_f32 (arg bounded in [-3, 0], no subnormal fixup needed).
// ---------------------------------------------------------------------------

typedef __attribute__((ext_vector_type(16))) _Float16 v16h;
typedef __attribute__((ext_vector_type(8)))  float    v8f;

#define NROWS 8192
#define DDIM  128
#define BSZ   4096
#define JCHUNKS 16
#define NTILES (NROWS / 16)              // 512 tiles per dimension
#define JT_PER_WAVE (NTILES / JCHUNKS)   // 32 j-tiles per wave (even)

union Frag16 { v16h h; unsigned u[8]; };

// --- pass 0: f32 -> f16 conversion, row norms, zero accumulators -----------
__global__ void hsic_convert(const float* __restrict__ feat,
                             _Float16* __restrict__ xh,
                             float* __restrict__ norms,
                             float* __restrict__ rowsum,
                             float* __restrict__ scalars) {
    __shared__ float red[DDIM];
    const int i = blockIdx.x;
    const int t = threadIdx.x;
    const float v = feat[(size_t)i * DDIM + t];
    xh[(size_t)i * DDIM + t] = (_Float16)v;
    red[t] = v * v;
    __syncthreads();
    for (int s = DDIM / 2; s > 0; s >>= 1) {
        if (t < s) red[t] += red[t + s];
        __syncthreads();
    }
    if (t == 0) { norms[i] = red[0]; rowsum[i] = 0.0f; }
    if (i == 0 && t < 2) scalars[t] = 0.0f;
}

// B fragment: 32x16 f16 per k-step, col n = lane&15 (columns are rows of X).
// VGPR v holds k-pair: k = ks*32 + half*16 + v*2  (32B contiguous per k-step).
__device__ __forceinline__ void load_btile(const _Float16* __restrict__ xh,
                                           int j0, int mlane, int half,
                                           Frag16* b) {
    const size_t brow = (size_t)(j0 + mlane) * DDIM;
#pragma unroll
    for (int ks = 0; ks < 4; ++ks) {
#pragma unroll
        for (int v = 0; v < 8; ++v) {
            const int k = ks * 32 + half * 16 + v * 2;
            b[ks].u[v] = *(const unsigned*)(xh + brow + k);
        }
    }
}

// Epilogue over one 16x16 f32 accumulator tile. C/D layout: VGPR r holds
// (row = i0 + r + 8*half, col = j0 + lane&15).
__device__ __forceinline__ void tile_epilogue(const v8f& c,
                                              const float* nrow, float ncol,
                                              bool posTile, int mlane, int half,
                                              float* rowAcc, float& sumK2,
                                              float& posK) {
    float kv[8];
#pragma unroll
    for (int r = 0; r < 8; ++r) {
        const float sq = fmaxf(nrow[r] + ncol - 2.0f * c[r], 0.0f);
        // exp(-sq/2) == exp2(-sq/(2*ln2)); sq <= ~4 so raw v_exp_f32 is safe.
        kv[r] = __builtin_amdgcn_exp2f(-0.72134752044448169f * sq);
        rowAcc[r] += kv[r];
        sumK2 += kv[r] * kv[r];
    }
    if (posTile) {  // wave-uniform; at most 1 of 32 tiles per wave
#pragma unroll
        for (int r = 0; r < 8; ++r)
            posK += ((r + half * 8) == mlane) ? kv[r] : 0.0f;
    }
}

// --- pass 1: tiled virtual-K reductions via WMMA ---------------------------
__global__ void __launch_bounds__(256)
hsic_gram(const _Float16* __restrict__ xh,
          const float* __restrict__ norms,
          float* __restrict__ rowsum,
          float* __restrict__ scalars) {
    const int lane   = threadIdx.x & 31;
    const int waveId = blockIdx.x * 8 + (threadIdx.x >> 5);
    const int itile  = waveId / JCHUNKS;       // 16-row strip owned by wave
    const int jc     = waveId % JCHUNKS;       // which chunk of j-tiles
    const int i0     = itile * 16;

    const int mlane = lane & 15;
    const int half  = lane >> 4;

    // A fragments: 16x32 f16 per k-step, row m = lane&15.
    // VGPR v holds k-pair: k = ks*32 + half*16 + v*2.
    Frag16 a[4];
    const size_t arow = (size_t)(i0 + mlane) * DDIM;
#pragma unroll
    for (int ks = 0; ks < 4; ++ks) {
#pragma unroll
        for (int v = 0; v < 8; ++v) {
            const int k = ks * 32 + half * 16 + v * 2;
            a[ks].u[v] = *(const unsigned*)(xh + arow + k);
        }
    }

    float nrow[8];
#pragma unroll
    for (int r = 0; r < 8; ++r) nrow[r] = norms[i0 + r + half * 8];

    float rowAcc[8];
#pragma unroll
    for (int r = 0; r < 8; ++r) rowAcc[r] = 0.0f;
    float sumK2 = 0.0f, posK = 0.0f;

    const int jtEnd = (jc + 1) * JT_PER_WAVE;
    for (int jt = jc * JT_PER_WAVE; jt < jtEnd; jt += 2) {
        const int j0a = jt * 16;
        const int j0b = j0a + 16;

        // Issue both tiles' loads (16x global_load_b128) before any wait so
        // tile b's loads overlap tile a's WMMAs + epilogue.
        Frag16 b0[4], b1[4];
        load_btile(xh, j0a, mlane, half, b0);
        load_btile(xh, j0b, mlane, half, b1);
        const float ncolA = norms[j0a + mlane];
        const float ncolB = norms[j0b + mlane];

        if (jt + 2 < NTILES)  // warm L0/L2 for the next pair
            __builtin_prefetch(xh + (size_t)((jt + 2) * 16 + mlane) * DDIM, 0, 1);

        v8f c0 = {};
#pragma unroll
        for (int ks = 0; ks < 4; ++ks)
            c0 = __builtin_amdgcn_wmma_f32_16x16x32_f16(
                     false, a[ks].h, false, b0[ks].h, (short)0, c0, false, false);
        tile_epilogue(c0, nrow, ncolA,
                      (i0 - j0a == BSZ) || (j0a - i0 == BSZ),
                      mlane, half, rowAcc, sumK2, posK);

        v8f c1 = {};
#pragma unroll
        for (int ks = 0; ks < 4; ++ks)
            c1 = __builtin_amdgcn_wmma_f32_16x16x32_f16(
                     false, a[ks].h, false, b1[ks].h, (short)0, c1, false, false);
        tile_epilogue(c1, nrow, ncolB,
                      (i0 - j0b == BSZ) || (j0b - i0 == BSZ),
                      mlane, half, rowAcc, sumK2, posK);
    }

    // Row sums: reduce across each 16-lane half (one half = one set of rows).
#pragma unroll
    for (int r = 0; r < 8; ++r) {
        float v = rowAcc[r];
        for (int m = 8; m > 0; m >>= 1) v += __shfl_xor(v, m, 16);
        if (mlane == 0) atomicAdd(&rowsum[i0 + r + half * 8], v);
    }
    // Scalars: full-wave reduce, one atomic per wave.
    for (int m = 16; m > 0; m >>= 1) {
        sumK2 += __shfl_xor(sumK2, m, 32);
        posK  += __shfl_xor(posK,  m, 32);
    }
    if (lane == 0) {
        atomicAdd(&scalars[0], sumK2);
        atomicAdd(&scalars[1], posK);
    }
}

// --- pass 2: final scalar ---------------------------------------------------
__global__ void hsic_finalize(const float* __restrict__ rowsum,
                              const float* __restrict__ scalars,
                              float* __restrict__ out) {
    __shared__ float s1[256], s2[256];
    const int t = threadIdx.x;
    float a = 0.0f, b = 0.0f;
    for (int i = t; i < NROWS; i += 256) {
        const float s = rowsum[i];
        a += s;
        b += s * s;
    }
    s1[t] = a; s2[t] = b;
    __syncthreads();
    for (int s = 128; s > 0; s >>= 1) {
        if (t < s) { s1[t] += s1[t + s]; s2[t] += s2[t + s]; }
        __syncthreads();
    }
    if (t == 0) {
        const float Nf = (float)NROWS;
        const float Bf = (float)BSZ;
        const float sumK  = s1[0];
        const float sumS2 = s2[0];
        const float sumK2 = scalars[0];
        const float pos   = scalars[1];

        const float offdiag = sumK - Nf;              // diag entries are exactly 1
        const float neg     = offdiag - pos;
        const float term1   = pos / (Bf * 2.0f);      // B*M*(M-1), M=2
        const float term2   = neg / (Bf * Bf * 4.0f); // B^2*M^2
        const float term3   = 1.0f;                   // 1/(M-1)
        const float hsic_zy = (Bf / (Bf - 1.0f)) * (term1 - term2 - term3);

        const float tm      = sumK / (Nf * Nf);
        const float hsic_zz = sumK2 / (Nf * Nf)
                            - 2.0f * sumS2 / (Nf * Nf * Nf)
                            + tm * tm;

        out[0] = -hsic_zy + 3.0f * sqrtf(fmaxf(hsic_zz, 0.0f));
    }
}

extern "C" void kernel_launch(void* const* d_in, const int* in_sizes, int n_in,
                              void* d_out, int out_size, void* d_ws, size_t ws_size,
                              hipStream_t stream) {
    (void)in_sizes; (void)n_in; (void)out_size; (void)ws_size;
    const float* feat = (const float*)d_in[0];

    // Workspace layout (~2.07 MB total):
    char*  ws      = (char*)d_ws;
    size_t off     = 0;
    _Float16* xh   = (_Float16*)(ws + off); off += (size_t)NROWS * DDIM * sizeof(_Float16);
    float* norms   = (float*)(ws + off);    off += (size_t)NROWS * sizeof(float);
    float* rowsum  = (float*)(ws + off);    off += (size_t)NROWS * sizeof(float);
    float* scalars = (float*)(ws + off);

    hsic_convert<<<NROWS, DDIM, 0, stream>>>(feat, xh, norms, rowsum, scalars);
    hsic_gram<<<(NTILES * JCHUNKS) / 8, 256, 0, stream>>>(xh, norms, rowsum, scalars);
    hsic_finalize<<<1, 256, 0, stream>>>(rowsum, scalars, (float*)d_out);
}
